// GraphNet_30064771072156
// MI455X (gfx1250) — matrix operand order repaired and checked by reading, not compiled, not run on hardware
//
#include <hip/hip_runtime.h>
#include <hip/hip_bf16.h>

#define NN 50000
#define NE 1600000

typedef __attribute__((ext_vector_type(16))) _Float16 v16h;
typedef __attribute__((ext_vector_type(8)))  float    v8f;

union AF { v16h v; _Float16 h[16]; uint4 q[2]; };
union CF { v8f  v; float f[8]; };

// ---------------- WMMA fragment helpers (CDNA5 wave32 layouts) ----------------

// A-matrix 16x32 f16: lane m=lane&15, kgroup=lane>>4.
// h[0..7]  = K = kb + kg*8 + 0..7   (contiguous -> one 16B load)
// h[8..15] = K = kb + 16 + kg*8 + 0..7
__device__ __forceinline__ AF load_afrag(const _Float16* base, int stride, int kb, int lane) {
    AF a;
    int m = lane & 15, kg = lane >> 4;
    const _Float16* p = base + m * stride + kb + kg * 8;
    a.q[0] = *(const uint4*)(p);
    a.q[1] = *(const uint4*)(p + 16);
    return a;
}

// B-matrix 32x16 f16 from transposed weights WT[n][k] (k contiguous):
// lane n=lane&15 (+ntile*16), kg=lane>>4; h[i] = W[kb + kg*8 + i][n] (i<8), W[kb + 16 + kg*8 + i-8][n] (i>=8)
__device__ __forceinline__ AF load_bfrag(const _Float16* wt, int kstride, int kb, int ntile, int lane) {
    AF b;
    int n = lane & 15, kg = lane >> 4;
    const _Float16* p = wt + (size_t)(ntile * 16 + n) * kstride + kb + kg * 8;
    b.q[0] = *(const uint4*)(p);
    b.q[1] = *(const uint4*)(p + 16);
    return b;
}

__device__ __forceinline__ void wmma_acc(CF& c, const AF& a, const AF& b) {
    c.v = __builtin_amdgcn_wmma_f32_16x16x32_f16(false, a.v, false, b.v,
                                                 (short)0, c.v, false, false);
}

__device__ __forceinline__ void relu2(CF c[2]) {
    for (int t = 0; t < 2; ++t)
        for (int r = 0; r < 8; ++r)
            c[t].f[r] = fmaxf(c[t].f[r], 0.0f);
}

// C layout: lane n0=lane&15 (col, +ntile*16), rows mb..mb+7 with mb=(lane>>4)*8
__device__ __forceinline__ void store_h(_Float16* H, CF c[2], int lane) {
    int n0 = lane & 15, mb = (lane >> 4) * 8;
    for (int t = 0; t < 2; ++t)
        for (int r = 0; r < 8; ++r)
            H[(mb + r) * 32 + t * 16 + n0] = (_Float16)c[t].f[r];
}

// 3-layer MLP on a 16-row tile: A(16 x kc0*32, LDS) -> relu -> 32 -> relu -> 32
__device__ __forceinline__ void run_mlp(const _Float16* A, int astride, int kc0,
                                        const _Float16* wt0, const _Float16* wt1,
                                        const _Float16* wt2, const float* bias,
                                        _Float16* H, int lane, CF c[2]) {
    int n0 = lane & 15;
    // layer 0
    for (int t = 0; t < 2; ++t) {
        float bv = bias[t * 16 + n0];
        for (int r = 0; r < 8; ++r) c[t].f[r] = bv;
    }
    for (int k = 0; k < kc0; ++k) {
        AF a = load_afrag(A, astride, k * 32, lane);
        for (int t = 0; t < 2; ++t) {
            AF b = load_bfrag(wt0, 96, k * 32, t, lane);
            wmma_acc(c[t], a, b);
        }
    }
    relu2(c);
    store_h(H, c, lane);
    __syncthreads();
    // layer 1
    CF d[2];
    for (int t = 0; t < 2; ++t) {
        float bv = bias[32 + t * 16 + n0];
        for (int r = 0; r < 8; ++r) d[t].f[r] = bv;
    }
    {
        AF a = load_afrag(H, 32, 0, lane);
        for (int t = 0; t < 2; ++t) {
            AF b = load_bfrag(wt1, 32, 0, t, lane);
            wmma_acc(d[t], a, b);
        }
    }
    relu2(d);
    __syncthreads();
    store_h(H, d, lane);
    __syncthreads();
    // layer 2 (no relu)
    for (int t = 0; t < 2; ++t) {
        float bv = bias[64 + t * 16 + n0];
        for (int r = 0; r < 8; ++r) c[t].f[r] = bv;
    }
    {
        AF a = load_afrag(H, 32, 0, lane);
        for (int t = 0; t < 2; ++t) {
            AF b = load_bfrag(wt2, 32, 0, t, lane);
            wmma_acc(c[t], a, b);
        }
    }
}

// ---------------- edge MLP + segment-sum scatter ----------------
// LAYER=1: input [x_r(1)|x_c(1)|w(1)] pad->96 ; LAYER=2/3: [x_r(32)|x_c(32)|eattr(32)]
template <int LAYER>
__global__ void __launch_bounds__(256) edge_kernel(
    const float* __restrict__ x_in, const _Float16* __restrict__ xcur,
    const int* __restrict__ ei, const float* __restrict__ ew,
    _Float16* __restrict__ eattr, float* __restrict__ s,
    const _Float16* wt0, const _Float16* wt1, const _Float16* wt2,
    const float* bias) {
    __shared__ _Float16 Ain[8][16 * 96];
    __shared__ _Float16 Hs[8][16 * 32];
    int lane = threadIdx.x & 31, w = threadIdx.x >> 5;
    long tile = (long)blockIdx.x * 8 + w;
    bool active = tile < (NE / 16);
    if (!active) tile = (NE / 16) - 1;
    long eb = tile * 16;
    const int* rowp = ei;
    const int* colp = ei + NE;
    _Float16* A = Ain[w];
    _Float16* H = Hs[w];

    if (LAYER == 1) {
        for (int idx = lane; idx < 192; idx += 32) ((uint4*)A)[idx] = make_uint4(0, 0, 0, 0);
        if (lane < 16) {
            long ge = eb + lane;
            A[lane * 96 + 0] = (_Float16)x_in[rowp[ge]];
            A[lane * 96 + 1] = (_Float16)x_in[colp[ge]];
            A[lane * 96 + 2] = (_Float16)ew[ge];
        }
    } else {
        for (int idx = lane; idx < 192; idx += 32) {  // 16 rows x 12 16B-chunks
            int r = idx / 12, c = idx % 12;
            long ge = eb + r;
            uint4 v;
            if (c < 4)      v = ((const uint4*)(xcur + (long)rowp[ge] * 32))[c];
            else if (c < 8) v = ((const uint4*)(xcur + (long)colp[ge] * 32))[c - 4];
            else            v = ((const uint4*)(eattr + ge * 32))[c - 8];
            ((uint4*)A)[r * 12 + c] = v;
        }
    }
    __syncthreads();

    CF c[2];
    run_mlp(A, 96, (LAYER == 1) ? 1 : 3, wt0, wt1, wt2, bias, H, lane, c);

    int n0 = lane & 15, mb = (lane >> 4) * 8;
    if (active) {
        for (int t = 0; t < 2; ++t)
            for (int r = 0; r < 8; ++r) {
                long ge = eb + mb + r;
                atomicAdd(s + (long)rowp[ge] * 32 + t * 16 + n0, c[t].f[r]);
            }
    }
    if (LAYER != 3) {  // stage e -> LDS -> vectorized contiguous global write
        __syncthreads();
        store_h(H, c, lane);
        __syncthreads();
        if (active) {
            uint4* src = (uint4*)H;                    // 64 x 16B
            uint4* dst = (uint4*)(eattr + eb * 32);    // 16 edges x 64B contiguous
            dst[lane] = src[lane];
            dst[lane + 32] = src[lane + 32];
        }
    }
}

// ---------------- node MLP: [x(<=32)|mean(32)] -> x_new ----------------
template <int LAYER>
__global__ void __launch_bounds__(256) node_kernel(
    const _Float16* __restrict__ xcur, const float* __restrict__ s,
    const float* __restrict__ cnt, _Float16* __restrict__ xnext,
    float* __restrict__ xfin,
    const _Float16* wt0, const _Float16* wt1, const _Float16* wt2,
    const float* bias) {
    __shared__ _Float16 Ain[8][16 * 64];
    __shared__ _Float16 Hs[8][16 * 32];
    int lane = threadIdx.x & 31, w = threadIdx.x >> 5;
    long tile = (long)blockIdx.x * 8 + w;
    bool active = tile < (NN / 16);
    if (!active) tile = (NN / 16) - 1;
    long nb = tile * 16;
    _Float16* A = Ain[w];
    _Float16* H = Hs[w];

    for (int idx = lane; idx < 64; idx += 32) {  // x half: 16 rows x 4 16B-chunks
        int r = idx >> 2, cc = idx & 3;
        ((uint4*)A)[r * 8 + cc] = ((const uint4*)(xcur + (nb + r) * 32))[cc];
    }
    for (int idx = lane; idx < 512; idx += 32) {  // mean half, scalar f32->f16
        int r = idx >> 5, j = idx & 31;
        long node = nb + r;
        float rcp = 1.0f / fmaxf(cnt[node], 1.0f);
        A[r * 64 + 32 + j] = (_Float16)(s[node * 32 + j] * rcp);
    }
    __syncthreads();

    CF c[2];
    run_mlp(A, 64, 2, wt0, wt1, wt2, bias, H, lane, c);

    int n0 = lane & 15, mb = (lane >> 4) * 8;
    if (active) {
        if (LAYER == 3) {
            if (n0 == 0)
                for (int r = 0; r < 8; ++r) xfin[nb + mb + r] = c[0].f[r];
        } else {
            for (int t = 0; t < 2; ++t)
                for (int r = 0; r < 8; ++r)
                    xnext[(nb + mb + r) * 32 + t * 16 + n0] = (_Float16)c[t].f[r];
        }
    }
}

// ---------------- small utility kernels ----------------
__global__ void zero_f32(float* p, long n) {
    long i = (long)blockIdx.x * blockDim.x + threadIdx.x;
    long st = (long)gridDim.x * blockDim.x;
    for (; i < n; i += st) p[i] = 0.0f;
}

__global__ void init_x(const float* __restrict__ x_in, _Float16* __restrict__ xA) {
    long i = (long)blockIdx.x * 256 + threadIdx.x;
    if (i < (long)NN * 32) {
        int j = i & 31;
        xA[i] = (j == 0) ? (_Float16)x_in[i >> 5] : (_Float16)0.0f;
    }
}

__global__ void count_kernel(const int* __restrict__ ei, float* __restrict__ cnt) {
    long e = (long)blockIdx.x * 256 + threadIdx.x;
    if (e < NE) atomicAdd(cnt + ei[e], 1.0f);
}

__global__ void final_kernel(const int* __restrict__ ei, const float* __restrict__ ew,
                             const float* __restrict__ xf, float* __restrict__ out) {
    long e = (long)blockIdx.x * 256 + threadIdx.x;
    if (e < NE) atomicAdd(out + ei[NE + e], ew[e] * xf[ei[e]]);
}

// Convert one MLP's weights to f16, transposed (N-major, K-contiguous), K-padded.
// WT0: 32n x 96k ; WT1/WT2: 32n x 32k ; bias: [b0|b1|b2] f32 x32 each (padded).
// xw==0 -> edge MLP (rows 0..k0-1 -> cols 0..k0-1)
// xw>0  -> node MLP (rows 0..xw-1 -> cols 0..xw-1 ; rows xw.. -> cols 32..)
__global__ void prep_mlp(const float* W0, const float* b0, const float* W1,
                         const float* b1, const float* W2, const float* b2,
                         int k0, int nc2, int xw,
                         _Float16* wt0, _Float16* wt1, _Float16* wt2, float* bias) {
    int tid = threadIdx.x;
    for (int i = tid; i < 32 * 96; i += 256) {
        int n = i / 96, k = i % 96;
        int sk = -1;
        if (xw == 0) { if (k < k0) sk = k; }
        else {
            if (k < xw) sk = k;
            else if (k >= 32 && k < 64) sk = xw + (k - 32);
        }
        wt0[n * 96 + k] = (sk >= 0) ? (_Float16)W0[sk * 32 + n] : (_Float16)0.0f;
    }
    for (int i = tid; i < 1024; i += 256) {
        int n = i / 32, k = i % 32;
        wt1[n * 32 + k] = (_Float16)W1[k * 32 + n];
    }
    for (int i = tid; i < 1024; i += 256) {
        int n = i / 32, k = i % 32;
        wt2[n * 32 + k] = (n < nc2) ? (_Float16)W2[k * nc2 + n] : (_Float16)0.0f;
    }
    for (int i = tid; i < 32; i += 256) {
        bias[i] = b0[i];
        bias[32 + i] = b1[i];
        bias[64 + i] = (i < nc2) ? b2[i] : 0.0f;
    }
}

// ---------------- host ----------------
extern "C" void kernel_launch(void* const* d_in, const int* in_sizes, int n_in,
                              void* d_out, int out_size, void* d_ws, size_t ws_size,
                              hipStream_t stream) {
    const float* x = (const float*)d_in[0];
    const int* ei = (const int*)d_in[1];
    const float* ew = (const float*)d_in[2];
    // d_in[3] = batch (unused)

    // Detect param flatten order. Insertion: [W0,b0,W1,b1,W2,b2] per group
    // (groups e1,n1,e2,n2,e3,n3) -> in_sizes[5]==32 (e1.b0).
    // Sorted pytree: groups e1,e2,e3,n1,n2,n3; tensors W0,W1,W2,b0,b1,b2
    // -> in_sizes[5]==1024 (e1.W1).
    bool sorted = (in_sizes[5] == 1024);
    static const int gmap[6] = {0, 3, 1, 4, 2, 5};  // logical e1,n1,e2,n2,e3,n3 -> sorted pos
    static const int tmap[6] = {0, 3, 1, 4, 2, 5};  // W0,b0,W1,b1,W2,b2 -> sorted pos
    auto P = [&](int g, int t) -> const float* {
        int idx = sorted ? (4 + gmap[g] * 6 + tmap[t]) : (4 + g * 6 + t);
        return (const float*)d_in[idx];
    };

    // workspace carve-up (256B aligned)
    char* base = (char*)d_ws;
    size_t off = 0;
    auto alloc = [&](size_t bytes) {
        size_t o = off;
        off = (off + bytes + 255) & ~(size_t)255;
        return (void*)(base + o);
    };
    _Float16* wt0[6]; _Float16* wt1[6]; _Float16* wt2[6]; float* bias[6];
    for (int m = 0; m < 6; ++m) wt0[m] = (_Float16*)alloc(32 * 96 * 2);
    for (int m = 0; m < 6; ++m) wt1[m] = (_Float16*)alloc(32 * 32 * 2);
    for (int m = 0; m < 6; ++m) wt2[m] = (_Float16*)alloc(32 * 32 * 2);
    for (int m = 0; m < 6; ++m) bias[m] = (float*)alloc(96 * 4);
    float* cnt = (float*)alloc((size_t)NN * 4);
    float* s = (float*)alloc((size_t)NN * 32 * 4);
    _Float16* xA = (_Float16*)alloc((size_t)NN * 32 * 2);
    _Float16* xB = (_Float16*)alloc((size_t)NN * 32 * 2);
    float* xfin = (float*)alloc((size_t)NN * 4);
    _Float16* eattr = (_Float16*)alloc((size_t)NE * 32 * 2);

    static const int K0[6] = {3, 33, 96, 64, 96, 64};
    static const int NC2[6] = {32, 32, 32, 32, 32, 1};
    static const int XW[6] = {0, 1, 0, 32, 0, 32};
    for (int m = 0; m < 6; ++m)
        prep_mlp<<<1, 256, 0, stream>>>(P(m, 0), P(m, 1), P(m, 2), P(m, 3), P(m, 4), P(m, 5),
                                        K0[m], NC2[m], XW[m], wt0[m], wt1[m], wt2[m], bias[m]);

    zero_f32<<<256, 256, 0, stream>>>(cnt, NN);
    init_x<<<(NN * 32 + 255) / 256, 256, 0, stream>>>(x, xA);
    count_kernel<<<(NE + 255) / 256, 256, 0, stream>>>(ei, cnt);

    int eg = NE / 128;                  // 12500 blocks, 8 wave-tiles each
    int ng = (NN / 16 + 7) / 8;         // 391 blocks

    // layer 1
    zero_f32<<<1024, 256, 0, stream>>>(s, (long)NN * 32);
    edge_kernel<1><<<eg, 256, 0, stream>>>(x, xA, ei, ew, eattr, s,
                                           wt0[0], wt1[0], wt2[0], bias[0]);
    node_kernel<1><<<ng, 256, 0, stream>>>(xA, s, cnt, xB, xfin,
                                           wt0[1], wt1[1], wt2[1], bias[1]);
    // layer 2
    zero_f32<<<1024, 256, 0, stream>>>(s, (long)NN * 32);
    edge_kernel<2><<<eg, 256, 0, stream>>>(x, xB, ei, ew, eattr, s,
                                           wt0[2], wt1[2], wt2[2], bias[2]);
    node_kernel<2><<<ng, 256, 0, stream>>>(xB, s, cnt, xA, xfin,
                                           wt0[3], wt1[3], wt2[3], bias[3]);
    // layer 3
    zero_f32<<<1024, 256, 0, stream>>>(s, (long)NN * 32);
    edge_kernel<3><<<eg, 256, 0, stream>>>(x, xA, ei, ew, eattr, s,
                                           wt0[4], wt1[4], wt2[4], bias[4]);
    node_kernel<3><<<ng, 256, 0, stream>>>(xA, s, cnt, xB, xfin,
                                           wt0[5], wt1[5], wt2[5], bias[5]);

    // final: out[col] += w * x3[row]
    zero_f32<<<256, 256, 0, stream>>>((float*)d_out, NN);
    final_kernel<<<(NE + 255) / 256, 256, 0, stream>>>(ei, ew, xfin, (float*)d_out);
}
